// MultimodalSentimentModel_35777077575723
// MI455X (gfx1250) — compile-verified, matrix operand order checked by reading, not run on hardware
//
#include <hip/hip_runtime.h>
#include <math.h>

typedef _Float16 half_t;
typedef __attribute__((ext_vector_type(8)))  _Float16 v8h;
typedef __attribute__((ext_vector_type(16))) _Float16 v16h;
typedef __attribute__((ext_vector_type(8)))  float    v8f;

// Flip to 0 if the assembler rejects the async-to-LDS mnemonic.
#define USE_ASYNC_LDS 1

// low 32 bits of a generic pointer into __shared__ == LDS byte offset
// (flat->LDS mapping truncates to addr[31:0], ISA 10.2)
__device__ inline unsigned lds_off(const void* p) {
  return (unsigned)(unsigned long long)p;
}

// ---------------------------------------------------------------------------
// Fragment loader for v_wmma_f32_16x16x32_f16 (works for global or LDS ptrs).
// 16-bit A-matrix 16x32 layout (ISA 7.12.2): lanes 0-15 hold K={0..7,16..23},
// lanes 16-31 hold K={8..15,24..31}; rows = lane&15. B mirrors with N across
// lanes (Bt stored row-major N x K so loads are contiguous).
// ---------------------------------------------------------------------------
__device__ inline v16h load_frag16(const half_t* __restrict__ rowk, int g) {
  const v8h lo = *(const v8h*)(rowk + g * 8);
  const v8h hi = *(const v8h*)(rowk + 16 + g * 8);
  v16h r;
#pragma unroll
  for (int i = 0; i < 8; ++i) { r[i] = lo[i]; r[8 + i] = hi[i]; }
  return r;
}

// ---------------------------------------------------------------------------
// LDS-staged, register-blocked WMMA GEMM:
//   C[m,n] = act(alpha * (sum_k A[m,k]*Bt[n,k] + bias[n]))
// Block = 256 threads = 8 waves (2 m-rows x 4 wave-cols); block tile 32 x 64*NT.
// Each wave owns NT 16x16 C tiles (one A fragment reused NT times per K-step).
// Per K-step the block stages a 32x32 A panel + (64*NT)x32 B panel into LDS
// with global_load_async_to_lds_b128 (ASYNCcnt), then computes from ds_load.
// M%32==0, N%(64*NT)==0, K%32==0 at every call site.
// ---------------------------------------------------------------------------
template <int NT>
__global__ __launch_bounds__(256)
void wmma_gemm_tpl(const half_t* __restrict__ A, int lda, long long sA,
                   const half_t* __restrict__ Bt, int ldb, long long sB,
                   float* __restrict__ Cf, half_t* __restrict__ Ch,
                   int ldc, long long sC,
                   const float* __restrict__ bias,
                   int K, int act, int accum,
                   const float* __restrict__ dynw, int dyncol) {
  constexpr int BN    = 64 * NT;   // block tile N
  constexpr int PITCH = 40;        // halves per staged row (32 data + 8 pad)
  constexpr int NCH   = 128 + BN * 4;  // 16B chunks per K-step (A:32rows, B:BNrows, 4/row)
  __shared__ half_t smA[32 * PITCH];
  __shared__ half_t smB[BN * PITCH];

  const int bz = blockIdx.z;
  A  += (long long)bz * sA;
  Bt += (long long)bz * sB;
  const long long coff = (long long)bz * sC;

  const int tid  = threadIdx.x;
  const int wid  = tid >> 5, lane = tid & 31;
  const int mr   = wid >> 2;            // wave m-row 0..1
  const int wc   = wid & 3;             // wave column 0..3
  const int mblk = blockIdx.x * 32;
  const int n0   = blockIdx.y * BN;
  const int g    = lane >> 4, l15 = lane & 15;

  v8f acc[NT];
#pragma unroll
  for (int nn = 0; nn < NT; ++nn) acc[nn] = (v8f){};

  for (int kk = 0; kk < K; kk += 32) {
    // ---- stage A/B K-panels into LDS (async DMA path) ----
    for (int c = tid; c < NCH; c += 256) {
      const half_t* gsrc;
      half_t* ldst;
      if (c < 128) {
        const int row = c >> 2, part = c & 3;
        gsrc = A + (long long)(mblk + row) * lda + kk + part * 8;
        ldst = smA + row * PITCH + part * 8;
      } else {
        const int cc = c - 128, row = cc >> 2, part = cc & 3;
        gsrc = Bt + (long long)(n0 + row) * ldb + kk + part * 8;
        ldst = smB + row * PITCH + part * 8;
      }
      if (kk + 32 < K) __builtin_prefetch(gsrc + 32, 0, 1);  // next K-panel -> global_prefetch_b8
#if USE_ASYNC_LDS
      asm volatile("global_load_async_to_lds_b128 %0, %1, off"
                   :: "v"(lds_off(ldst)), "v"(gsrc) : "memory");
#else
      *(v8h*)ldst = *(const v8h*)gsrc;
#endif
    }
#if USE_ASYNC_LDS
    asm volatile("s_wait_asynccnt 0" ::: "memory");
#endif
    __syncthreads();

    // ---- compute: one A fragment feeds NT WMMAs ----
    const v16h a = load_frag16(smA + (mr * 16 + l15) * PITCH, g);
#pragma unroll
    for (int nn = 0; nn < NT; ++nn) {
      const v16h bb = load_frag16(smB + (wc * 16 + nn * 64 + l15) * PITCH, g);
      acc[nn] = __builtin_amdgcn_wmma_f32_16x16x32_f16(false, a, false, bb,
                                                       (short)0, acc[nn], false, false);
    }
    __syncthreads();   // protect LDS panels from next stage (WAR)
  }

  float alpha = 1.0f;
  if (dynw) alpha = dynw[dyncol] + dynw[3 + dyncol] + dynw[6 + dyncol];
  const int m0    = mblk + mr * 16;
  const int rbase = m0 + g * 8;   // C tile: VGPR r -> row rbase+r, col = lane&15
#pragma unroll
  for (int nn = 0; nn < NT; ++nn) {
    const int col  = n0 + wc * 16 + nn * 64 + l15;
    const float bv = bias ? bias[col] : 0.0f;
#pragma unroll
    for (int r = 0; r < 8; ++r) {
      float v = (acc[nn][r] + bv) * alpha;
      const long long idx = coff + (long long)(rbase + r) * ldc + col;
      if (accum) v += Cf[idx];
      if (act == 1) v = v > 0.0f ? v : 0.0f;
      if (Cf) Cf[idx] = v;
      if (Ch) Ch[idx] = (half_t)v;
    }
  }
}

// ---------------------------------------------------------------------------
// Attention scores over the batch axis (L=16): one wave per (n,h) computes the
// full 16x16 scores tile with WMMA; K-loop over dh (global fragment loads).
// qkv layout: row (l*Nn + n), pitch cols; q at h*dh, k at koff + h*dh.
// scores layout: [((n*NH + h)*16 + l)*16 + m]
// ---------------------------------------------------------------------------
__global__ __launch_bounds__(256)
void attn_scores_kernel(const half_t* __restrict__ qkv, int pitch, int koff,
                        int dh, float scale, float* __restrict__ scores,
                        int Nn, int NH) {
  const int wid  = (int)((blockIdx.x * blockDim.x + threadIdx.x) >> 5);
  const int lane = threadIdx.x & 31;
  const int h = wid % NH, n = wid / NH;
  if (n >= Nn) return;                     // wave-uniform
  const int g = lane >> 4, l15 = lane & 15;

  const half_t* arow = qkv + ((long long)l15 * Nn + n) * pitch + h * dh;        // q row l
  const half_t* brow = qkv + ((long long)l15 * Nn + n) * pitch + koff + h * dh; // k row m

  v8f acc = {};
  for (int kk = 0; kk < dh; kk += 32) {
    v16h a = load_frag16(arow + kk, g);
    v16h bb = load_frag16(brow + kk, g);
    acc = __builtin_amdgcn_wmma_f32_16x16x32_f16(false, a, false, bb,
                                                 (short)0, acc, false, false);
  }
  float* base = scores + (long long)(n * NH + h) * 256;
#pragma unroll
  for (int r = 0; r < 8; ++r) {
    const int l = g * 8 + r;   // query index (C row)
    const int m = l15;         // key index (C col)
    base[l * 16 + m] = acc[r] * scale;
  }
}

// softmax over the 16-wide key axis; one thread per (n,h,l) row
__global__ void softmax16_kernel(const float* __restrict__ s,
                                 float* __restrict__ o, int rows) {
  const int t = blockIdx.x * blockDim.x + threadIdx.x;
  if (t >= rows) return;
  const float* r = s + (long long)t * 16;
  float mx = r[0];
#pragma unroll
  for (int i = 1; i < 16; ++i) mx = fmaxf(mx, r[i]);
  float sum = 0.0f, e[16];
#pragma unroll
  for (int i = 0; i < 16; ++i) { e[i] = expf(r[i] - mx); sum += e[i]; }
  const float inv = 1.0f / sum;
  float* w = o + (long long)t * 16;
#pragma unroll
  for (int i = 0; i < 16; ++i) w[i] = e[i] * inv;
}

// o[l,n,c] = sum_m attn[n,h,l,m] * v[m,n,c]   (K=16: VALU dot, ~1 GFLOP total)
__global__ void attn_o_kernel(const float* __restrict__ attn,
                              const half_t* __restrict__ qkv, int pitch,
                              int voff, int dh, half_t* __restrict__ o,
                              int opitch, int Nn, int NH, long long total) {
  const long long t = (long long)blockIdx.x * blockDim.x + threadIdx.x;
  if (t >= total) return;
  const int c = (int)(t % opitch);
  const long long row = t / opitch;      // l*Nn + n
  const int n = (int)(row % Nn);
  const int l = (int)(row / Nn);
  const int h = c / dh;
  const float* aw = attn + ((long long)(n * NH + h) * 16 + l) * 16;
  float s = 0.0f;
#pragma unroll
  for (int m = 0; m < 16; ++m)
    s += aw[m] * (float)qkv[((long long)m * Nn + n) * pitch + voff + c];
  o[t] = (half_t)s;
}

// es/ed attention logits + normalized rows for cosine graph
__global__ void gat_vec_kernel(const float* __restrict__ h,
                               const float* __restrict__ dist,
                               const float* __restrict__ asrc,
                               const float* __restrict__ adst,
                               float* __restrict__ es, float* __restrict__ ed,
                               half_t* __restrict__ nrmh, int T) {
  const int t = blockIdx.x * blockDim.x + threadIdx.x;
  if (t >= T) return;
  const float* hr = h + (long long)t * 128;
  const float* dr = dist + (long long)t * 128;
  float a = 0.f, b = 0.f, nn = 0.f;
  for (int d = 0; d < 128; ++d) {
    a += hr[d] * asrc[d]; b += hr[d] * adst[d]; nn += dr[d] * dr[d];
  }
  es[t] = a; ed[t] = b;
  const float inv = 1.0f / fmaxf(sqrtf(nn), 1e-12f);
  for (int d = 0; d < 128; ++d) nrmh[(long long)t * 128 + d] = (half_t)(dr[d] * inv);
}

// masked softmax over sources i for target (b,j); writes alphaT[b,j,i] (f16)
// adjacency: (i==j) || (i<j && cos>0.9); cos is symmetric so read row j.
__global__ __launch_bounds__(256)
void gat_alpha_kernel(const half_t* __restrict__ cosm,
                      const float* __restrict__ es, const float* __restrict__ ed,
                      half_t* __restrict__ alphaT, int S) {
  const int b = blockIdx.y, j = blockIdx.x, tid = threadIdx.x;
  __shared__ float red[256];
  const float edj = ed[b * S + j];
  const half_t* crow = cosm + ((long long)b * S + j) * S;
  float vals[2], mx = -3.4e38f;
#pragma unroll
  for (int q = 0; q < 2; ++q) {
    const int i = tid + q * 256;
    const bool adj = (i == j) || (i < j && (float)crow[i] > 0.9f);
    float e = es[b * S + i] + edj;
    e = e > 0.f ? e : 0.2f * e;            // leaky_relu(.,0.2)
    vals[q] = adj ? e : -1e9f;
    mx = fmaxf(mx, vals[q]);
  }
  red[tid] = mx; __syncthreads();
  for (int s2 = 128; s2 > 0; s2 >>= 1) {
    if (tid < s2) red[tid] = fmaxf(red[tid], red[tid + s2]);
    __syncthreads();
  }
  mx = red[0]; __syncthreads();
  float ex[2], sum = 0.f;
#pragma unroll
  for (int q = 0; q < 2; ++q) { ex[q] = expf(vals[q] - mx); sum += ex[q]; }
  red[tid] = sum; __syncthreads();
  for (int s2 = 128; s2 > 0; s2 >>= 1) {
    if (tid < s2) red[tid] += red[tid + s2];
    __syncthreads();
  }
  const float inv = 1.0f / red[0];
#pragma unroll
  for (int q = 0; q < 2; ++q)
    alphaT[((long long)b * S + j) * S + tid + q * 256] = (half_t)(ex[q] * inv);
}

// per-batch transpose h (S x 128 f32) -> ht (128 x S f16)
__global__ void htrans_kernel(const float* __restrict__ h, half_t* __restrict__ ht,
                              int S, int D, long long total) {
  const long long t = (long long)blockIdx.x * blockDim.x + threadIdx.x;
  if (t >= total) return;
  const int b = (int)(t / ((long long)S * D));
  const int rem = (int)(t % ((long long)S * D));
  const int d = rem / S, i = rem % S;
  ht[t] = (half_t)h[((long long)b * S + i) * D + d];
}

__global__ void route_kernel(const half_t* __restrict__ rf1,
                             const float* __restrict__ w, const float* __restrict__ bb,
                             float* __restrict__ route_out, float* __restrict__ route_ws,
                             int T) {
  const int t = blockIdx.x * blockDim.x + threadIdx.x;
  if (t >= T) return;
  float s = bb[0];
  for (int d = 0; d < 128; ++d) s += (float)rf1[(long long)t * 128 + d] * w[d];
  const float r = 1.0f / (1.0f + expf(-s));
  route_out[t] = r; route_ws[t] = r;
}

__global__ void combine_kernel(const float* __restrict__ route,
                               const float* __restrict__ formal,
                               const float* __restrict__ informal,
                               float* __restrict__ out, long long total) {
  const long long t = (long long)blockIdx.x * blockDim.x + threadIdx.x;
  if (t >= total) return;
  const float r = route[t / 128];
  out[t] = r * formal[t] + (1.0f - r) * informal[t];
}

__global__ void cvt_f16_kernel(const float* __restrict__ s, half_t* __restrict__ d,
                               long long n) {
  const long long t = (long long)blockIdx.x * blockDim.x + threadIdx.x;
  if (t < n) d[t] = (half_t)s[t];
}

// W (K x N f32, row-major) -> Wt (N x K f16, row-major)
__global__ void transw_kernel(const float* __restrict__ w, half_t* __restrict__ wt,
                              int K, int N, long long total) {
  const long long t = (long long)blockIdx.x * blockDim.x + threadIdx.x;
  if (t >= total) return;
  const int n = (int)(t / K), k = (int)(t % K);
  wt[t] = (half_t)w[(long long)k * N + n];
}

// ---------------------------------------------------------------------------
// Host orchestration
// ---------------------------------------------------------------------------
extern "C" void kernel_launch(void* const* d_in, const int* in_sizes, int n_in,
                              void* d_out, int out_size, void* d_ws, size_t ws_size,
                              hipStream_t stream) {
  (void)in_sizes; (void)n_in; (void)out_size; (void)ws_size;
  const int Bn = 16, S = 512, D = 128, CTX = 2560, HID = 512;
  const int T = Bn * S;                        // 8192 tokens
  float* out = (float*)d_out;

  // ---- inputs ----
  const float* x0   = (const float*)d_in[0];
  const float* x1   = (const float*)d_in[1];
  const float* x2   = (const float*)d_in[2];
  const float* ctx  = (const float*)d_in[3];
  const float* sW[3]  = {(const float*)d_in[4],  (const float*)d_in[10], (const float*)d_in[16]};
  const float* sb[3]  = {(const float*)d_in[5],  (const float*)d_in[11], (const float*)d_in[17]};
  const float* pW[3]  = {(const float*)d_in[6],  (const float*)d_in[12], (const float*)d_in[18]};
  const float* pb[3]  = {(const float*)d_in[7],  (const float*)d_in[13], (const float*)d_in[19]};
  const float* dW[3]  = {(const float*)d_in[8],  (const float*)d_in[14], (const float*)d_in[20]};
  const float* db[3]  = {(const float*)d_in[9],  (const float*)d_in[15], (const float*)d_in[21]};
  const float* dynw   = (const float*)d_in[22];
  const float* doW    = (const float*)d_in[23];  const float* dob   = (const float*)d_in[24];
  const float* gatW   = (const float*)d_in[25];
  const float* att_src= (const float*)d_in[26];  const float* att_dst=(const float*)d_in[27];
  const float* gat_b  = (const float*)d_in[28];
  const float* rqkvw  = (const float*)d_in[29];  const float* rqkvb = (const float*)d_in[30];
  const float* routw  = (const float*)d_in[31];  const float* routb = (const float*)d_in[32];
  const float* rf1w   = (const float*)d_in[33];  const float* rf1b  = (const float*)d_in[34];
  const float* rf2w   = (const float*)d_in[35];  const float* rf2b  = (const float*)d_in[36];
  const float* fqkvw  = (const float*)d_in[37];  const float* fqkvb = (const float*)d_in[38];
  const float* foutw  = (const float*)d_in[39];  const float* foutb = (const float*)d_in[40];
  const float* ff1w   = (const float*)d_in[41];  const float* ff1b  = (const float*)d_in[42];
  const float* ff2w   = (const float*)d_in[43];  const float* ff2b  = (const float*)d_in[44];
  const float* if1w   = (const float*)d_in[45];  const float* if1b  = (const float*)d_in[46];
  const float* if2w   = (const float*)d_in[47];  const float* if2b  = (const float*)d_in[48];

  // ---- d_out slice offsets (floats) ----
  const long long P = (long long)T * D;
  float* o_out   = out;
  float* o_route = out + P;
  float* o_sh[3] = {out + P + T, out + P + T + P, out + P + T + 2 * P};
  float* o_pr[3] = {out + P + T + 3 * P, out + P + T + 4 * P, out + P + T + 5 * P};
  float* o_dist  = out + P + T + 6 * P;

  // ---- workspace bump allocator ----
  size_t off = 0;
  auto alloc = [&](size_t bytes) -> void* {
    off = (off + 255) & ~(size_t)255;
    void* p = (char*)d_ws + off;
    off += bytes;
    return p;
  };
  auto ah = [&](long long n) { return (half_t*)alloc((size_t)n * 2); };
  auto af = [&](long long n) { return (float*)alloc((size_t)n * 4); };

  const int FD[3] = {512, 1024, 1024};
  half_t* xh[3];
  xh[0] = ah((long long)T * 512);
  xh[1] = ah((long long)T * 1024);
  xh[2] = ah((long long)T * 1024);
  half_t* ctxh = ah((long long)T * CTX);
  half_t *sWt[3], *pWt[3], *dWt[3];
  for (int m = 0; m < 3; ++m) {
    sWt[m] = ah((long long)D * FD[m]);
    pWt[m] = ah((long long)D * FD[m]);
    dWt[m] = ah((long long)D * D);
  }
  half_t* doWt  = ah((long long)D * D);
  half_t* gatWt = ah((long long)D * D);
  half_t* rqkvT = ah((long long)3 * CTX * CTX);
  half_t* routT = ah((long long)CTX * CTX);
  half_t* rf1T  = ah((long long)D * CTX);
  half_t* fqkvT = ah((long long)3 * D * D);
  half_t* foutT = ah((long long)D * D);
  half_t* ff1T  = ah((long long)HID * D);
  half_t* ff2T  = ah((long long)D * HID);
  half_t* if1T  = ah((long long)HID * D);
  half_t* if2T  = ah((long long)D * HID);

  half_t* sharedh[3] = {ah(P), ah(P), ah(P)};
  float*  pre_f  = af(P);
  half_t* preh   = ah(P);
  half_t* disth  = ah(P);
  float*  h_f    = af(P);
  half_t* cosm   = ah((long long)Bn * S * S);
  float*  es     = af(T);
  float*  ed     = af(T);
  half_t* nrmh   = ah(P);
  half_t* alphaT = ah((long long)Bn * S * S);
  half_t* ht     = ah((long long)Bn * D * S);
  half_t* gath   = ah(P);
  half_t* qkvh   = ah((long long)T * 3 * CTX);
  float*  rsc    = af((long long)S * 4 * 256);
  float*  ratt   = af((long long)S * 4 * 256);
  half_t* oh     = ah((long long)T * CTX);
  half_t* rh     = ah((long long)T * CTX);
  half_t* rf1h   = ah(P);
  half_t* fqkvh  = ah((long long)T * 3 * D);
  float*  fsc    = af((long long)S * 4 * 256);
  float*  fatt   = af((long long)S * 4 * 256);
  half_t* foh    = ah(P);
  half_t* fh     = ah(P);
  half_t* f1h    = ah((long long)T * HID);
  float*  formal = af(P);
  half_t* ihh    = ah((long long)T * HID);
  float*  informal = af(P);
  float*  route_ws = af(T);

  auto cvt = [&](const float* s, half_t* d, long long n) {
    cvt_f16_kernel<<<(unsigned)((n + 255) / 256), 256, 0, stream>>>(s, d, n);
  };
  auto trw = [&](const float* w, half_t* wt, int K, int N) {
    long long n = (long long)K * N;
    transw_kernel<<<(unsigned)((n + 255) / 256), 256, 0, stream>>>(w, wt, K, N, n);
  };
  auto gemm = [&](const half_t* A, int lda, long long sA,
                  const half_t* Bt, int ldb, long long sB,
                  float* Cf, half_t* Ch, int ldc, long long sC,
                  const float* bias, int M, int N, int K,
                  int act, int accum, const float* dw, int dcol, int nb) {
    const int nt = (N % 256 == 0) ? 4 : (N % 128 == 0) ? 2 : 1;
    dim3 g((unsigned)(M / 32), (unsigned)(N / (64 * nt)), (unsigned)nb);
    if (nt == 4)
      wmma_gemm_tpl<4><<<g, 256, 0, stream>>>(A, lda, sA, Bt, ldb, sB, Cf, Ch,
                                              ldc, sC, bias, K, act, accum, dw, dcol);
    else if (nt == 2)
      wmma_gemm_tpl<2><<<g, 256, 0, stream>>>(A, lda, sA, Bt, ldb, sB, Cf, Ch,
                                              ldc, sC, bias, K, act, accum, dw, dcol);
    else
      wmma_gemm_tpl<1><<<g, 256, 0, stream>>>(A, lda, sA, Bt, ldb, sB, Cf, Ch,
                                              ldc, sC, bias, K, act, accum, dw, dcol);
  };

  // ---- phase A: precision conversion ----
  cvt(x0, xh[0], (long long)T * 512);
  cvt(x1, xh[1], (long long)T * 1024);
  cvt(x2, xh[2], (long long)T * 1024);
  cvt(ctx, ctxh, (long long)T * CTX);
  for (int m = 0; m < 3; ++m) {
    trw(sW[m], sWt[m], FD[m], D);
    trw(pW[m], pWt[m], FD[m], D);
    trw(dW[m], dWt[m], D, D);
  }
  trw(doW, doWt, D, D);       trw(gatW, gatWt, D, D);
  trw(rqkvw, rqkvT, CTX, 3 * CTX);
  trw(routw, routT, CTX, CTX);
  trw(rf1w, rf1T, CTX, D);
  trw(fqkvw, fqkvT, D, 3 * D);
  trw(foutw, foutT, D, D);
  trw(ff1w, ff1T, D, HID);    trw(ff2w, ff2T, HID, D);
  trw(if1w, if1T, D, HID);    trw(if2w, if2T, HID, D);

  // ---- phase B: feature decoupler ----
  for (int m = 0; m < 3; ++m) {
    gemm(xh[m], FD[m], 0, sWt[m], FD[m], 0, o_sh[m], sharedh[m], D, 0,
         sb[m], T, D, FD[m], 0, 0, nullptr, 0, 1);
    gemm(xh[m], FD[m], 0, pWt[m], FD[m], 0, o_pr[m], nullptr, D, 0,
         pb[m], T, D, FD[m], 0, 0, nullptr, 0, 1);
  }

  // ---- phase C: dynamic graph distiller (fused weighted projection sum) ----
  for (int m = 0; m < 3; ++m)
    gemm(sharedh[m], D, 0, dWt[m], D, 0, pre_f, preh, D, 0,
         db[m], T, D, D, 0, m > 0 ? 1 : 0, dynw, m, 1);
  gemm(preh, D, 0, doWt, D, 0, o_dist, disth, D, 0, dob, T, D, D, 0, 0, nullptr, 0, 1);

  // ---- phase D: GAT on cosine graph ----
  gemm(disth, D, 0, gatWt, D, 0, h_f, nullptr, D, 0, nullptr, T, D, D, 0, 0, nullptr, 0, 1);
  gat_vec_kernel<<<(T + 255) / 256, 256, 0, stream>>>(h_f, o_dist, att_src, att_dst,
                                                      es, ed, nrmh, T);
  gemm(nrmh, D, (long long)S * D, nrmh, D, (long long)S * D,
       nullptr, cosm, S, (long long)S * S, nullptr, S, S, D, 0, 0, nullptr, 0, Bn);
  { dim3 g((unsigned)S, (unsigned)Bn, 1);
    gat_alpha_kernel<<<g, 256, 0, stream>>>(cosm, es, ed, alphaT, S); }
  { long long n = (long long)Bn * D * S;
    htrans_kernel<<<(unsigned)((n + 255) / 256), 256, 0, stream>>>(h_f, ht, S, D, n); }
  gemm(alphaT, S, (long long)S * S, ht, S, (long long)D * S,
       nullptr, gath, D, (long long)S * D, gat_b, S, D, S, 1, 0, nullptr, 0, Bn);

  // ---- phase E: router MHA (L=16 over batch axis, E=2560, nh=4, dh=640) ----
  gemm(ctxh, CTX, 0, rqkvT, CTX, 0, nullptr, qkvh, 3 * CTX, 0,
       rqkvb, T, 3 * CTX, CTX, 0, 0, nullptr, 0, 1);
  attn_scores_kernel<<<(S * 4 * 32) / 256, 256, 0, stream>>>(
      qkvh, 3 * CTX, CTX, 640, 1.0f / sqrtf(640.0f), rsc, S, 4);
  softmax16_kernel<<<(S * 4 * 16 + 255) / 256, 256, 0, stream>>>(rsc, ratt, S * 4 * 16);
  { long long n = (long long)T * CTX;
    attn_o_kernel<<<(unsigned)((n + 255) / 256), 256, 0, stream>>>(
        ratt, qkvh, 3 * CTX, 2 * CTX, 640, oh, CTX, S, 4, n); }
  gemm(oh, CTX, 0, routT, CTX, 0, nullptr, rh, CTX, 0, routb, T, CTX, CTX, 0, 0, nullptr, 0, 1);
  gemm(rh, CTX, 0, rf1T, CTX, 0, nullptr, rf1h, D, 0, rf1b, T, D, CTX, 1, 0, nullptr, 0, 1);
  route_kernel<<<(T + 255) / 256, 256, 0, stream>>>(rf1h, rf2w, rf2b, o_route, route_ws, T);

  // ---- phase F: formal expert MHA (E=128, nh=4, dh=32) + FFN ----
  gemm(gath, D, 0, fqkvT, D, 0, nullptr, fqkvh, 3 * D, 0,
       fqkvb, T, 3 * D, D, 0, 0, nullptr, 0, 1);
  attn_scores_kernel<<<(S * 4 * 32) / 256, 256, 0, stream>>>(
      fqkvh, 3 * D, D, 32, 1.0f / sqrtf(32.0f), fsc, S, 4);
  softmax16_kernel<<<(S * 4 * 16 + 255) / 256, 256, 0, stream>>>(fsc, fatt, S * 4 * 16);
  { long long n = (long long)T * D;
    attn_o_kernel<<<(unsigned)((n + 255) / 256), 256, 0, stream>>>(
        fatt, fqkvh, 3 * D, 2 * D, 32, foh, D, S, 4, n); }
  gemm(foh, D, 0, foutT, D, 0, nullptr, fh, D, 0, foutb, T, D, D, 0, 0, nullptr, 0, 1);
  gemm(fh, D, 0, ff1T, D, 0, nullptr, f1h, HID, 0, ff1b, T, HID, D, 1, 0, nullptr, 0, 1);
  gemm(f1h, HID, 0, ff2T, HID, 0, formal, nullptr, D, 0, ff2b, T, D, HID, 0, 0, nullptr, 0, 1);

  // ---- informal expert + gated combine ----
  gemm(gath, D, 0, if1T, D, 0, nullptr, ihh, HID, 0, if1b, T, HID, D, 1, 0, nullptr, 0, 1);
  gemm(ihh, HID, 0, if2T, HID, 0, informal, nullptr, D, 0, if2b, T, D, HID, 0, 0, nullptr, 0, 1);
  { long long n = (long long)T * D;
    combine_kernel<<<(unsigned)((n + 255) / 256), 256, 0, stream>>>(
        route_ws, formal, informal, o_out, n); }
}
